// MultiScaleAttention_23845658427825
// MI455X (gfx1250) — compile-verified, matrix operand order checked by reading, not run on hardware
//
#include <hip/hip_runtime.h>
#include <hip/hip_bf16.h>
#include <math.h>
#include <stdint.h>

// ---------------------------------------------------------------------------
// MultiScaleAttention (MViT window attention w/ dwconv pooling) for gfx1250.
// - All GEMMs (qkv proj, Q.K^T, rel-pos bias, P.V, out proj) via
//   v_wmma_f32_16x16x32_bf16; all LDS fragments contiguous (ds_load_b128).
// - All intermediates bf16 in global memory (halves HBM traffic).
// - Attention Q/K window staging via gfx1250 async LDS copies
//   (global_load_async_to_lds_b128 + s_wait_asynccnt).
// - Projection GEMMs use 128x128 tiles (8 waves) to halve L2 operand traffic.
// ---------------------------------------------------------------------------

typedef __attribute__((ext_vector_type(16))) __bf16 v16bf;
typedef __attribute__((ext_vector_type(8)))  __bf16 v8bf;
typedef __attribute__((ext_vector_type(4)))  __bf16 v4bf;
typedef __attribute__((ext_vector_type(8)))  float  v8f;

#define NUM_HEADS 12
#define DIM       768
#define HEAD_DIM  64
#define WSZ       14
#define LWIN      196      // WSZ*WSZ
#define LPAD      208      // 13*16
#define LPAD2     224      // 7*32
#define KT_TILES  13
#define IMGW      56
#define NPIX      3136
#define NHEADS96  96
#define M_TOK     25088

// LDS byte offsets inside the dynamic shared block (attention kernel).
#define QB_OFF   0u                          // [224][64] bf16
#define KB_OFF   (QB_OFF + 224u * 64u * 2u)  // [208][64] bf16
#define VT_OFF   (KB_OFF + 208u * 64u * 2u)  // [64][224] bf16 (V^T)
#define PB_OFF   (VT_OFF + 64u * 224u * 2u)  // [4][16][224] bf16
#define RHB_OFF  (PB_OFF + 4u * 16u * 224u * 2u)  // [27][64] bf16
#define RWB_OFF  (RHB_OFF + 27u * 64u * 2u)       // [27][64] bf16
#define REL_OFF  (RWB_OFF + 27u * 64u * 2u)       // [208][28] f32
#define SMEM_TOTAL (REL_OFF + 208u * 28u * 4u)

__device__ __forceinline__ v8f wmma_bf16(v16bf a, v16bf b, v8f c) {
  return __builtin_amdgcn_wmma_f32_16x16x32_bf16(
      false, a, false, b, (short)0, c, false, false);
}

// bf16 16x32 fragment, explicit per-lane row (row-major, contiguous K).
__device__ __forceinline__ v16bf frag_row(const __bf16* S, int ld, int row,
                                          int cbase, int lane) {
  v16bf f;
  const __bf16* p = S + (size_t)row * ld + cbase + ((lane >> 4) * 8);
#pragma unroll
  for (int v = 0; v < 8; ++v) {
    int k0 = ((v & 3) * 2) + ((v >> 2) * 16);
    f[2 * v]     = p[k0];
    f[2 * v + 1] = p[k0 + 1];
  }
  return f;
}

__device__ __forceinline__ v16bf frag_rm(const __bf16* S, int ld, int row_base,
                                         int cbase, int lane) {
  return frag_row(S, ld, row_base + (lane & 15), cbase, lane);
}

__device__ __forceinline__ v4bf pack4(float a, float b, float c, float d) {
  v4bf t;
  t[0] = (__bf16)a; t[1] = (__bf16)b; t[2] = (__bf16)c; t[3] = (__bf16)d;
  return t;
}

// gfx1250 async LDS copy: 16B per active lane, global -> LDS, ASYNCcnt.
__device__ __forceinline__ void async_copy16(unsigned lds_off, const void* gptr) {
  unsigned long long ga = (unsigned long long)(size_t)gptr;
  asm volatile("global_load_async_to_lds_b128 %0, %1, off"
               :: "v"(lds_off), "v"(ga) : "memory");
}
__device__ __forceinline__ void wait_async0() {
  asm volatile("s_wait_asynccnt 0" ::: "memory");
}

// ---------------------------------------------------------------------------
// GEMM: C[M,N] = A[M,K] * B[K,N](f32) + bias.  TA in {float, __bf16},
// TC in {float, __bf16}. 128x128 tile, K step 32, 8 waves (4 along M x 2
// along N; each wave 32x64 = 2x4 WMMA tiles). Bt staged transposed.
// ---------------------------------------------------------------------------
template <typename TA, typename TC>
__global__ void __launch_bounds__(256)
gemm_any(const TA* __restrict__ A, const float* __restrict__ B,
         const float* __restrict__ bias, TC* __restrict__ C,
         int M, int N, int K) {
  __shared__ __bf16 As[128 * 40];
  __shared__ __bf16 Bt[128 * 40];
  const int tid  = threadIdx.x;
  const int lane = tid & 31;
  const int wave = tid >> 5;          // 0..7
  const int m0 = blockIdx.x * 128;
  const int n0 = blockIdx.y * 128;
  const int wm = (wave & 3) * 32;     // 4 waves along M
  const int wn = (wave >> 2) * 64;    // 2 waves along N

  v8f acc[2][4];
#pragma unroll
  for (int i = 0; i < 2; ++i)
#pragma unroll
    for (int j = 0; j < 4; ++j)
      acc[i][j] = v8f{0.f, 0.f, 0.f, 0.f, 0.f, 0.f, 0.f, 0.f};

  for (int k0 = 0; k0 < K; k0 += 32) {
    if (k0 + 32 < K) {  // speculative prefetch of next K tile
      __builtin_prefetch(&A[(size_t)(m0 + (tid >> 1)) * K + k0 + 32 + (tid & 1) * 16], 0, 1);
      __builtin_prefetch(&B[(size_t)(k0 + 32 + (tid >> 7)) * N + n0 + (tid & 127)], 0, 1);
    }
    if constexpr (sizeof(TA) == 4) {     // f32 -> bf16 convert-stage
      for (int i = tid; i < 128 * 8; i += 256) {
        int r = i >> 3, kg = (i & 7) * 4;
        const float4 f = *(const float4*)((const float*)A + (size_t)(m0 + r) * K + k0 + kg);
        *(v4bf*)(As + r * 40 + kg) = pack4(f.x, f.y, f.z, f.w);
      }
    } else {                             // bf16 -> straight 16B copies
      for (int i = tid; i < 128 * 4; i += 256) {
        int r = i >> 2, kg = (i & 3) * 8;
        *(uint4*)(As + r * 40 + kg) =
            *(const uint4*)((const __bf16*)A + (size_t)(m0 + r) * K + k0 + kg);
      }
    }
    // B transposed: 128 cols x 8 k-groups of 4 (weights are f32)
    for (int i = tid; i < 128 * 8; i += 256) {
      int c = i & 127, kg = (i >> 7) * 4;
      v4bf t;
#pragma unroll
      for (int j = 0; j < 4; ++j)
        t[j] = (__bf16)B[(size_t)(k0 + kg + j) * N + n0 + c];
      *(v4bf*)(Bt + c * 40 + kg) = t;
    }
    __syncthreads();
    v16bf a0 = frag_rm(As, 40, wm + 0, 0, lane);
    v16bf a1 = frag_rm(As, 40, wm + 16, 0, lane);
#pragma unroll
    for (int j = 0; j < 4; ++j) {
      v16bf bf = frag_rm(Bt, 40, wn + j * 16, 0, lane);
      acc[0][j] = wmma_bf16(a0, bf, acc[0][j]);
      acc[1][j] = wmma_bf16(a1, bf, acc[1][j]);
    }
    __syncthreads();
  }

  const int nl = lane & 15;
  const int mo = (lane >> 4) * 8;
#pragma unroll
  for (int i = 0; i < 2; ++i)
#pragma unroll
    for (int j = 0; j < 4; ++j) {
      int n = n0 + wn + j * 16 + nl;
      float bv = bias[n];
#pragma unroll
      for (int r = 0; r < 8; ++r) {
        int m = m0 + wm + i * 16 + r + mo;
        C[(size_t)m * N + n] = (TC)(acc[i][j][r] + bv);
      }
    }
}

// ---------------------------------------------------------------------------
// Depthwise 3x3 conv (pad 1, groups=64) + LayerNorm(64). bf16 in/out, f32 math.
// ---------------------------------------------------------------------------
__global__ void __launch_bounds__(128)
dwconv_ln(const __bf16* __restrict__ qkv,
          const float* __restrict__ wq, const float* __restrict__ wk,
          const float* __restrict__ wv,
          const float* __restrict__ gq, const float* __restrict__ bq,
          const float* __restrict__ gk, const float* __restrict__ bk,
          const float* __restrict__ gv, const float* __restrict__ bv,
          __bf16* __restrict__ outln) {
  const int lane = threadIdx.x & 31;
  const int wave = threadIdx.x >> 5;
  const int p = blockIdx.x * 4 + wave;
  const int y = p / IMGW, x = p % IMGW;
  const int n96 = blockIdx.y, which = blockIdx.z;
  const int img = n96 / NUM_HEADS, head = n96 % NUM_HEADS;
  const float* w  = (which == 0) ? wq : (which == 1) ? wk : wv;
  const float* g  = (which == 0) ? gq : (which == 1) ? gk : gv;
  const float* bb = (which == 0) ? bq : (which == 1) ? bk : bv;
  const size_t cbase = (size_t)img * NPIX * (3 * DIM) + (size_t)which * DIM
                     + (size_t)head * HEAD_DIM;
  float a0 = 0.f, a1 = 0.f;
#pragma unroll
  for (int dy = -1; dy <= 1; ++dy) {
    int yy = y + dy;
    if ((unsigned)yy >= IMGW) continue;
#pragma unroll
    for (int dx = -1; dx <= 1; ++dx) {
      int xx = x + dx;
      if ((unsigned)xx >= IMGW) continue;
      const __bf16* src = qkv + cbase + (size_t)(yy * IMGW + xx) * (3 * DIM);
      int wi = ((dy + 1) * 3 + (dx + 1)) * HEAD_DIM;
      a0 += (float)src[lane]      * w[wi + lane];
      a1 += (float)src[lane + 32] * w[wi + lane + 32];
    }
  }
  float s = a0 + a1;
#pragma unroll
  for (int off = 16; off > 0; off >>= 1) s += __shfl_xor(s, off, 32);
  float mean = s * (1.f / 64.f);
  float d0 = a0 - mean, d1 = a1 - mean;
  float vs = d0 * d0 + d1 * d1;
#pragma unroll
  for (int off = 16; off > 0; off >>= 1) vs += __shfl_xor(vs, off, 32);
  float inv = rsqrtf(vs * (1.f / 64.f) + 1e-5f);
  __bf16* dst = outln + ((size_t)(which * NHEADS96 + n96) * NPIX + p) * HEAD_DIM;
  dst[lane]      = (__bf16)(d0 * inv * g[lane]      + bb[lane]);
  dst[lane + 32] = (__bf16)(d1 * inv * g[lane + 32] + bb[lane + 32]);
}

// ---------------------------------------------------------------------------
// Window attention, one block (4 waves) per window.
// ---------------------------------------------------------------------------
__global__ void __launch_bounds__(128)
attn_win(const __bf16* __restrict__ qln, const __bf16* __restrict__ kln,
         const __bf16* __restrict__ vln, const float* __restrict__ rph,
         const float* __restrict__ rpw, __bf16* __restrict__ outb) {
  extern __shared__ char smem[];
  __bf16* qb  = (__bf16*)(smem + QB_OFF);
  __bf16* kb  = (__bf16*)(smem + KB_OFF);
  __bf16* vt  = (__bf16*)(smem + VT_OFF);
  __bf16* pb  = (__bf16*)(smem + PB_OFF);
  __bf16* rhb = (__bf16*)(smem + RHB_OFF);
  __bf16* rwb = (__bf16*)(smem + RWB_OFF);
  float*  rel = (float*)(smem + REL_OFF);

  const int tid = threadIdx.x, lane = tid & 31, wave = tid >> 5;
  const int b = blockIdx.x;
  const int n96 = b >> 4, wh = (b >> 2) & 3, ww = b & 3;
  const int img = n96 / NUM_HEADS, head = n96 % NUM_HEADS;
  const size_t base = (size_t)n96 * NPIX * HEAD_DIM;
  const v4bf z4 = pack4(0.f, 0.f, 0.f, 0.f);

  // ---- Phase 0a: zero pad regions (DS stores, disjoint from async dests) --
  for (int i = tid; i < 4 * 16 * (LPAD2 / 4); i += 128) ((v4bf*)pb)[i] = z4;
  for (int i = tid; i < 28 * 16; i += 128)               // qb rows 196..223
    *(v4bf*)(qb + (196 + (i >> 4)) * 64 + (i & 15) * 4) = z4;
  for (int i = tid; i < 12 * 16; i += 128)               // kb rows 196..207
    *(v4bf*)(kb + (196 + (i >> 4)) * 64 + (i & 15) * 4) = z4;
  for (int i = tid; i < 12 * 28; i += 128) rel[196 * 28 + i] = 0.f;

  // ---- Phase 0b: async Q/K window copy (bf16 -> bf16, 16B per lane) -------
  for (int i = tid; i < LWIN * 8; i += 128) {
    int row = i >> 3, cg = (i & 7) * 8;
    int gp = (wh * WSZ + row / WSZ) * IMGW + (ww * WSZ + row % WSZ);
    size_t go = base + (size_t)gp * HEAD_DIM + cg;
    unsigned lo = (unsigned)(row * 64 + cg) * 2u;
    async_copy16(QB_OFF + lo, qln + go);
    async_copy16(KB_OFF + lo, kln + go);
  }

  // ---- Phase 0c: V^T staging and rel tables (manual) ----------------------
  for (int i = tid; i < 64 * (LPAD2 / 4); i += 128) {
    int c = i & 63, kg = i >> 6;
    v4bf v4 = z4;
    if (kg * 4 < LWIN) {   // keys 4kg..4kg+3 all < 196 (196 = 49*4)
#pragma unroll
      for (int j = 0; j < 4; ++j) {
        int key = kg * 4 + j;
        int gp = (wh * WSZ + key / WSZ) * IMGW + (ww * WSZ + key % WSZ);
        v4[j] = vln[base + (size_t)gp * HEAD_DIM + c];
      }
    }
    *(v4bf*)(vt + c * LPAD2 + kg * 4) = v4;
  }
  for (int i = tid; i < 27 * 16; i += 128) {
    int row = i >> 4, cg = (i & 15) * 4;
    const float4 hf = *(const float4*)(rph + (size_t)row * HEAD_DIM + cg);
    const float4 wf = *(const float4*)(rpw + (size_t)row * HEAD_DIM + cg);
    *(v4bf*)(rhb + row * 64 + cg) = pack4(hf.x, hf.y, hf.z, hf.w);
    *(v4bf*)(rwb + row * 64 + cg) = pack4(wf.x, wf.y, wf.z, wf.w);
  }
  wait_async0();
  __syncthreads();

  const int nl = lane & 15;
  const int mo = (lane >> 4) * 8;

  // ---- Phase 1: decomposed rel-pos bias via WMMA (split H/W paths) --------
  for (int g = wave; g < 28; g += 4) {
    if (g < 14) {
      const int qh = g;
      int trow = qh + 13 - nl; if (trow < 0) trow = 0;
      v8f acc = v8f{0.f, 0.f, 0.f, 0.f, 0.f, 0.f, 0.f, 0.f};
      acc = wmma_bf16(frag_row(qb, 64, qh * 14 + nl, 0, lane),
                      frag_row(rhb, 64, trow, 0, lane), acc);
      acc = wmma_bf16(frag_row(qb, 64, qh * 14 + nl, 32, lane),
                      frag_row(rhb, 64, trow, 32, lane), acc);
      if (nl < 14) {
#pragma unroll
        for (int r = 0; r < 8; ++r) {
          int m = r + mo;
          if (m < 14) rel[(qh * 14 + m) * 28 + nl] = acc[r];
        }
      }
    } else {
      const int qw = g - 14;
      int trow = qw + 13 - nl; if (trow < 0) trow = 0;
      v8f acc = v8f{0.f, 0.f, 0.f, 0.f, 0.f, 0.f, 0.f, 0.f};
      acc = wmma_bf16(frag_row(qb, 64, qw + 14 * nl, 0, lane),
                      frag_row(rwb, 64, trow, 0, lane), acc);
      acc = wmma_bf16(frag_row(qb, 64, qw + 14 * nl, 32, lane),
                      frag_row(rwb, 64, trow, 32, lane), acc);
      if (nl < 14) {
#pragma unroll
        for (int r = 0; r < 8; ++r) {
          int m = r + mo;
          if (m < 14) rel[(qw + 14 * m) * 28 + 14 + nl] = acc[r];
        }
      }
    }
  }
  __syncthreads();

  // ---- Phase 2: attention per q-tile --------------------------------------
  __bf16* pw = pb + wave * 16 * LPAD2;

  for (int qt = wave; qt < KT_TILES; qt += 4) {
    const int row0 = qt * 16;
    v16bf qa0 = frag_rm(qb, 64, row0, 0, lane);
    v16bf qa1 = frag_rm(qb, 64, row0, 32, lane);

    float sreg[KT_TILES][8];
#pragma unroll
    for (int kt = 0; kt < KT_TILES; ++kt) {
      v8f acc = v8f{0.f, 0.f, 0.f, 0.f, 0.f, 0.f, 0.f, 0.f};
      acc = wmma_bf16(qa0, frag_rm(kb, 64, kt * 16, 0, lane), acc);
      acc = wmma_bf16(qa1, frag_rm(kb, 64, kt * 16, 32, lane), acc);
      int kcol = kt * 16 + nl;
      if (kcol < LWIN) {
        int kh = kcol / WSZ, kw = kcol % WSZ;
#pragma unroll
        for (int r = 0; r < 8; ++r) {
          int qrow = row0 + r + mo;   // rel rows 196..207 zeroed
          sreg[kt][r] = acc[r] * 0.125f + rel[qrow * 28 + kh]
                      + rel[qrow * 28 + 14 + kw];
        }
      } else {
#pragma unroll
        for (int r = 0; r < 8; ++r) sreg[kt][r] = -1e30f;
      }
    }

    // row softmax across the 16 lanes of a row (xor<16 keeps halves apart)
    float rmax[8], rsum[8];
#pragma unroll
    for (int r = 0; r < 8; ++r) rmax[r] = sreg[0][r];
#pragma unroll
    for (int kt = 1; kt < KT_TILES; ++kt)
#pragma unroll
      for (int r = 0; r < 8; ++r) rmax[r] = fmaxf(rmax[r], sreg[kt][r]);
#pragma unroll
    for (int r = 0; r < 8; ++r) {
#pragma unroll
      for (int off = 8; off > 0; off >>= 1)
        rmax[r] = fmaxf(rmax[r], __shfl_xor(rmax[r], off, 32));
      rsum[r] = 0.f;
    }
#pragma unroll
    for (int kt = 0; kt < KT_TILES; ++kt)
#pragma unroll
      for (int r = 0; r < 8; ++r) {
        float pe = __expf(sreg[kt][r] - rmax[r]);
        sreg[kt][r] = pe;
        rsum[r] += pe;
      }
#pragma unroll
    for (int r = 0; r < 8; ++r) {
#pragma unroll
      for (int off = 8; off > 0; off >>= 1) rsum[r] += __shfl_xor(rsum[r], off, 32);
      rsum[r] = 1.f / rsum[r];
    }

    // stage normalized P (bf16): pw[m][key]
#pragma unroll
    for (int kt = 0; kt < KT_TILES; ++kt)
#pragma unroll
      for (int r = 0; r < 8; ++r)
        pw[(size_t)(r + mo) * LPAD2 + kt * 16 + nl] =
            (__bf16)(sreg[kt][r] * rsum[r]);

    // (P@V) computed transposed: D[c][m] = V^T . P^T; both operands contiguous.
    v8f oacc[4];
#pragma unroll
    for (int ct = 0; ct < 4; ++ct)
      oacc[ct] = v8f{0.f, 0.f, 0.f, 0.f, 0.f, 0.f, 0.f, 0.f};
#pragma unroll
    for (int kk = 0; kk < 7; ++kk) {
      v16bf pbf = frag_rm(pw, LPAD2, 0, kk * 32, lane);        // B: lane -> query m
#pragma unroll
      for (int ct = 0; ct < 4; ++ct) {
        v16bf va = frag_rm(vt, LPAD2, ct * 16, kk * 32, lane); // A: lane -> channel
        oacc[ct] = wmma_bf16(va, pbf, oacc[ct]);
      }
    }

    // epilogue: C rows = channel, cols = query row; lane stores 8 contiguous
    // channels per tile as one 16B bf16 store, residual fused.
    int qrow = row0 + nl;
    if (qrow < LWIN) {
      int gp = (wh * WSZ + qrow / WSZ) * IMGW + (ww * WSZ + qrow % WSZ);
      const size_t rbase = base + (size_t)gp * HEAD_DIM;
      const size_t obase = ((size_t)img * NPIX + gp) * DIM + head * HEAD_DIM;
#pragma unroll
      for (int ct = 0; ct < 4; ++ct) {
        int c0 = ct * 16 + mo;
        v8bf res = *(const v8bf*)(qln + rbase + c0);
        v8bf o;
#pragma unroll
        for (int r = 0; r < 8; ++r)
          o[r] = (__bf16)(oacc[ct][r] + (float)res[r]);
        *(v8bf*)(outb + obase + c0) = o;
      }
    }
  }
}

// ---------------------------------------------------------------------------
// Host-side launch
// ---------------------------------------------------------------------------
extern "C" void kernel_launch(void* const* d_in, const int* in_sizes, int n_in,
                              void* d_out, int out_size, void* d_ws, size_t ws_size,
                              hipStream_t stream) {
  (void)in_sizes; (void)n_in; (void)out_size; (void)ws_size;
  const float* x      = (const float*)d_in[0];
  const float* w_qkv  = (const float*)d_in[1];
  const float* b_qkv  = (const float*)d_in[2];
  const float* w_proj = (const float*)d_in[3];
  const float* b_proj = (const float*)d_in[4];
  const float* pqw = (const float*)d_in[5];
  const float* pkw = (const float*)d_in[6];
  const float* pvw = (const float*)d_in[7];
  const float* gq = (const float*)d_in[8];  const float* bq = (const float*)d_in[9];
  const float* gk = (const float*)d_in[10]; const float* bk = (const float*)d_in[11];
  const float* gv = (const float*)d_in[12]; const float* bv = (const float*)d_in[13];
  const float* rph = (const float*)d_in[14];
  const float* rpw = (const float*)d_in[15];

  // workspace (all bf16): qkv_raw [25088][2304], qln [3][96][3136][64],
  // outb [25088][768]
  __bf16* qkv_raw = (__bf16*)d_ws;
  __bf16* qln     = qkv_raw + (size_t)M_TOK * 3 * DIM;
  __bf16* outb    = qln + (size_t)3 * NHEADS96 * NPIX * HEAD_DIM;

  // 1) qkv projection (f32 in, bf16 out), 128x128 tiles
  gemm_any<float, __bf16><<<dim3(M_TOK / 128, (3 * DIM) / 128), 256, 0, stream>>>(
      x, w_qkv, b_qkv, qkv_raw, M_TOK, 3 * DIM, DIM);

  // 2) depthwise conv + layernorm for q/k/v (bf16 in/out)
  dwconv_ln<<<dim3(NPIX / 4, NHEADS96, 3), 128, 0, stream>>>(
      qkv_raw, pqw, pkw, pvw, gq, bq, gk, bk, gv, bv, qln);

  // 3) fused window attention (bf16 in/out)
  const __bf16* qpost = qln;
  const __bf16* kpost = qln + (size_t)NHEADS96 * NPIX * HEAD_DIM;
  const __bf16* vpost = kpost + (size_t)NHEADS96 * NPIX * HEAD_DIM;
  attn_win<<<dim3(1536), 128, (size_t)SMEM_TOTAL, stream>>>(
      qpost, kpost, vpost, rph, rpw, outb);

  // 4) output projection (bf16 in, f32 out), 128x128 tiles
  gemm_any<__bf16, float><<<dim3(M_TOK / 128, DIM / 128), 256, 0, stream>>>(
      outb, w_proj, b_proj, (float*)d_out, M_TOK, DIM, DIM);
}
